// FCNN_heteroBessel_41798621724980
// MI455X (gfx1250) — matrix operand order repaired and checked by reading, not compiled, not run on hardware
//
#include <hip/hip_runtime.h>

typedef __attribute__((ext_vector_type(16))) _Float16 v16h;
typedef __attribute__((ext_vector_type(8)))  float    v8f;

#define TWO_OVER_PI 0.636619772f

// ---------- branch-free sin/cos for |x| < ~1e4 (Cody-Waite + minimax) ----------
__device__ __forceinline__ void sincos_red(float x, float& s, float& c) {
    float n = rintf(x * 0.636619772f);                 // round(x * 2/pi)
    float r = fmaf(-n, 1.57079637e+00f, x);            // x - n*float(pi/2)
    r       = fmaf(-n, -4.37113900e-08f, r);           // - n*(pi/2 - float(pi/2))
    int  qm = ((int)n) & 3;
    float z = r * r;
    float sp = fmaf(z, fmaf(z, -1.9515296e-4f, 8.3321609e-3f), -1.6666655e-1f);
    float sr = fmaf(sp * z, r, r);                     // sin(r)
    float cp = fmaf(z, fmaf(z, 2.4433157e-5f, -1.3887316e-3f), 4.1666646e-2f);
    float cr = fmaf(cp, z * z, fmaf(-0.5f, z, 1.0f));  // cos(r)
    float ss = (qm & 1) ? cr : sr;
    float cc = (qm & 1) ? sr : cr;
    ss = (qm >= 2) ? -ss : ss;
    cc = (qm == 1 || qm == 2) ? -cc : cc;
    s = ss; c = cc;
}

__device__ __forceinline__ float sin_red(float x) {
    float s, c;
    sincos_red(x, s, c);
    return s;
}

// ---------- log(x) for x >= 1 (finite, normal) ----------
__device__ __forceinline__ float log_fast(float x) {
    int xi = __float_as_int(x);
    int e  = (xi >> 23) - 127;
    float m = __int_as_float((xi & 0x007fffff) | 0x3f800000);   // [1,2)
    if (m > 1.41421356f) { m *= 0.5f; e += 1; }                 // [sqrt2/2, sqrt2)
    float f = m - 1.0f;
    float u = f / (2.0f + f);
    float z = u * u;
    float p = fmaf(z, fmaf(z, fmaf(z, 0.14285715f, 0.2f), 0.33333334f), 1.0f);
    return fmaf((float)e, 0.693147181f, 2.0f * u * p);
}

// ---------- fused J0 & J1 (NR coefficients, matching the reference) ----------
__device__ __forceinline__ void j01_small(float x, float& j0, float& j1) {  // 0 <= x < 8
    float y = x * x;
    float p0 = 57568490574.0f + y * (-13362590354.0f + y * (651619640.7f +
               y * (-11214424.18f + y * (77392.33017f + y * (-184.9052456f)))));
    float q0 = 57568490411.0f + y * (1029532985.0f + y * (9494680.718f +
               y * (59272.64853f + y * (267.8532712f + y))));
    float p1 = x * (72362614232.0f + y * (-7895059235.0f + y * (242396853.1f +
               y * (-2972611.439f + y * (15704.4826f + y * (-30.16036606f))))));
    float q1 = 144725228442.0f + y * (2300535178.0f + y * (18583304.74f +
               y * (99447.43394f + y * (376.9991397f + y))));
    j0 = p0 / q0;
    j1 = p1 / q1;
}

// asymptotic P/Q polynomials, shared by J and Y of same order
__device__ __forceinline__ void asym_pq(float y, float& P0, float& Q0, float& P1, float& Q1) {
    P0 = 1.0f + y * (-0.001098628627f + y * (2.734510407e-05f +
         y * (-2.073370639e-06f + y * 2.093887211e-07f)));
    Q0 = -0.01562499995f + y * (0.0001430488765f + y * (-6.911147651e-06f +
         y * (7.621095161e-07f + y * (-9.34935152e-08f))));
    P1 = 1.0f + y * (0.00183105f + y * (-3.516396496e-05f +
         y * (2.457520174e-06f + y * (-2.40337019e-07f))));
    Q1 = 0.04687499995f + y * (-0.0002002690873f + y * (8.449199096e-06f +
         y * (-8.8228987e-07f + y * 1.05787412e-07f)));
}

// ---------- J2 or J4, one code copy (noinline keeps the I$ footprint tiny) ----------
__device__ __noinline__ float j_even(float xin, int want4) {
    float x = fabsf(xin);
    float res;
    if (x < 0.5f) {                                    // Taylor: recurrence unstable near 0
        float x2 = x * x, u = 0.25f * x2;
        if (!want4)
            res = (0.125f * x2) * (1.0f + u * (-(1.0f / 3.0f) + u * ((1.0f / 24.0f) - u * (1.0f / 360.0f))));
        else
            res = (x2 * x2 * (1.0f / 384.0f)) * (1.0f + u * (-0.2f + u * ((1.0f / 60.0f) - u * (1.0f / 1260.0f))));
    } else {
        float j0, j1;
        if (x < 8.0f) {
            j01_small(x, j0, j1);
        } else {
            float z = 8.0f / x, y = z * z;
            float P0, Q0, P1, Q1;
            asym_pq(y, P0, Q0, P1, Q1);
            float s, c;
            sincos_red(x - 0.785398164f, s, c);        // xx0; xx1 = xx0 - pi/2
            float fac = sqrtf(TWO_OVER_PI / x);
            j0 = fac * (c * P0 - z * s * Q0);
            j1 = fac * (s * P1 + z * c * Q1);          // cos(xx1)=s, sin(xx1)=-c
        }
        float inv = 1.0f / x;
        float j2 = 2.0f * inv * j1 - j0;
        if (!want4) {
            res = j2;
        } else {
            float j3 = 4.0f * inv * j2 - j1;
            res = 6.0f * inv * j3 - j2;
        }
    }
    return res;
}

// ---------- Y2 or Y4 with the reference's max(x,1) clamp and NaN->0 ----------
__device__ __noinline__ float y_even(float xin, int want4) {
    float x = fmaxf(xin, 1.0f);
    float y0, y1;
    if (x < 8.0f) {
        float j0, j1;
        j01_small(x, j0, j1);
        float lg = log_fast(x);
        float y = x * x;
        float p0 = -2957821389.0f + y * (7062834065.0f + y * (-512359803.6f +
                   y * (10879881.29f + y * (-86327.92757f + y * 228.4622733f))));
        float q0 = 40076544269.0f + y * (745249964.8f + y * (7189466.438f +
                   y * (47447.2647f + y * (226.1030244f + y))));
        float p1 = x * (-4.900604943e13f + y * (1.27527439e13f + y * (-5.153438139e11f +
                   y * (7.349264551e9f + y * (-4.237922726e7f + y * 8.511937935e4f)))));
        float q1 = 2.49958057e14f + y * (4.244419664e12f + y * (3.733650367e10f +
                   y * (2.245904002e8f + y * (1.02042605e6f + y * (3.549632885e3f + y)))));
        y0 = p0 / q0 + TWO_OVER_PI * j0 * lg;
        y1 = p1 / q1 + TWO_OVER_PI * (j1 * lg - 1.0f / x);
    } else {
        float z = 8.0f / x, y = z * z;
        float P0, Q0, P1, Q1;
        asym_pq(y, P0, Q0, P1, Q1);
        float s, c;
        sincos_red(x - 0.785398164f, s, c);            // xx0; xx1 = xx0 - pi/2
        float fac = sqrtf(TWO_OVER_PI / x);
        y0 = fac * (s * P0 + z * c * Q0);
        y1 = fac * (-c * P1 + z * s * Q1);             // sin(xx1)=-c, cos(xx1)=s
    }
    float inv = 1.0f / x;
    float y2 = 2.0f * inv * y1 - y0;
    float res;
    if (!want4) {
        res = y2;
    } else {
        float y3 = 4.0f * inv * y2 - y1;
        res = 6.0f * inv * y3 - y2;
    }
    if (res != res) res = 0.0f;
    return res;
}

// ---------- pack av[] -> f16 hi/lo fragments and accumulate 3 WMMAs ----------
__device__ __forceinline__ void mma_chunk(const float av[16], int c, int khiB,
                                          const float* s_ow, v8f& acc) {
    v16h ah, al, bh, bl;
    #pragma unroll
    for (int e = 0; e < 16; ++e) {
        _Float16 h = (_Float16)av[e];
        ah[e] = h;
        al[e] = (_Float16)(av[e] - (float)h);
    }
    const int kb = c * 32 + khiB;
    #pragma unroll
    for (int e = 0; e < 16; ++e) {
        float w = s_ow[kb + e + ((e >= 8) ? 8 : 0)];   // B[k][n] = out_w[k] for all n
        _Float16 h = (_Float16)w;
        bh[e] = h;
        bl[e] = (_Float16)(w - (float)h);
    }
    // (ah+al)*(bh+bl) ~= ah*bh + ah*bl + al*bh  (f32 accumulate)
    acc = __builtin_amdgcn_wmma_f32_16x16x32_f16(false, ah, false, bh, (short)0, acc, false, false);
    acc = __builtin_amdgcn_wmma_f32_16x16x32_f16(false, ah, false, bl, (short)0, acc, false, false);
    acc = __builtin_amdgcn_wmma_f32_16x16x32_f16(false, al, false, bh, (short)0, acc, false, false);
}

// One wave = 16 rows; 512 features as 16 chunks of K=32 via
// v_wmma_f32_16x16x32_f16 with hi/lo f16 splitting (emulated-f32 GEMV).
__global__ __launch_bounds__(256) void FCNN_heteroBessel_kernel(
    const float* __restrict__ t,
    const float* __restrict__ sin_w, const float* __restrict__ sin_b,
    const float* __restrict__ j2w,   const float* __restrict__ j2b,
    const float* __restrict__ j4w,   const float* __restrict__ j4b,
    const float* __restrict__ y2w,   const float* __restrict__ y2b,
    const float* __restrict__ y4w,   const float* __restrict__ y4b,
    const float* __restrict__ ow,    const float* __restrict__ ob,
    float* __restrict__ out)
{
    __shared__ float s_w[512];
    __shared__ float s_b[512];
    __shared__ float s_ow[512];

    const int tid = threadIdx.x;
    for (int f = tid; f < 512; f += 256) {
        float w, b;
        if (f < 256)      { w = sin_w[f];     b = sin_b[f];     }
        else if (f < 320) { w = j2w[f - 256]; b = j2b[f - 256]; }
        else if (f < 384) { w = j4w[f - 320]; b = j4b[f - 320]; }
        else if (f < 448) { w = y2w[f - 384]; b = y2b[f - 384]; }
        else              { w = y4w[f - 448]; b = y4b[f - 448]; }
        s_w[f] = w; s_b[f] = b; s_ow[f] = ow[f];
    }
    __syncthreads();

    const int lane  = tid & 31;
    const int wave  = tid >> 5;
    const int rbase = (blockIdx.x * 8 + wave) * 16;
    const int m     = lane & 15;              // this lane's matrix row
    const float rr  = t[2 * (rbase + m)];     // r
    const float th  = t[2 * (rbase + m) + 1]; // theta
    const float bias = ob[0];

    // A-matrix f16 16x32: lane = M; half e -> K = e + (e>=8?8:0) + (lane>=16?8:0)
    const int khiA = (lane >= 16) ? 8 : 0;
    // B-matrix f16 32x16: lane = N; half e -> K = e + (e>=8?8:0) + (lane>=16?8:0)
    const int khiB = (lane >= 16) ? 8 : 0;

    v8f acc = {};

    // -------- phase 1: sin chunks (c = 0..7) --------
    #pragma unroll 1
    for (int c = 0; c < 8; ++c) {
        const int fbase = c * 32 + khiA;
        float av[16];
        #pragma unroll
        for (int e = 0; e < 16; ++e) {
            const int f = fbase + e + ((e >= 8) ? 8 : 0);
            av[e] = sin_red(fmaf(th, s_w[f], s_b[f]));
        }
        mma_chunk(av, c, khiB, s_ow, acc);
    }
    // -------- phase 2: J2 chunks (c = 8..9) --------
    #pragma unroll 1
    for (int c = 8; c < 10; ++c) {
        const int fbase = c * 32 + khiA;
        float av[16];
        #pragma unroll
        for (int e = 0; e < 16; ++e) {
            const int f = fbase + e + ((e >= 8) ? 8 : 0);
            av[e] = j_even(fmaf(rr, s_w[f], s_b[f]), 0);
        }
        mma_chunk(av, c, khiB, s_ow, acc);
    }
    // -------- phase 3: J4 chunks (c = 10..11) --------
    #pragma unroll 1
    for (int c = 10; c < 12; ++c) {
        const int fbase = c * 32 + khiA;
        float av[16];
        #pragma unroll
        for (int e = 0; e < 16; ++e) {
            const int f = fbase + e + ((e >= 8) ? 8 : 0);
            av[e] = j_even(fmaf(rr, s_w[f], s_b[f]), 1);
        }
        mma_chunk(av, c, khiB, s_ow, acc);
    }
    // -------- phase 4: Y2 chunks (c = 12..13) --------
    #pragma unroll 1
    for (int c = 12; c < 14; ++c) {
        const int fbase = c * 32 + khiA;
        float av[16];
        #pragma unroll
        for (int e = 0; e < 16; ++e) {
            const int f = fbase + e + ((e >= 8) ? 8 : 0);
            av[e] = y_even(fmaf(rr, s_w[f], s_b[f]), 0);
        }
        mma_chunk(av, c, khiB, s_ow, acc);
    }
    // -------- phase 5: Y4 chunks (c = 14..15) --------
    #pragma unroll 1
    for (int c = 14; c < 16; ++c) {
        const int fbase = c * 32 + khiA;
        float av[16];
        #pragma unroll
        for (int e = 0; e < 16; ++e) {
            const int f = fbase + e + ((e >= 8) ? 8 : 0);
            av[e] = y_even(fmaf(rr, s_w[f], s_b[f]), 1);
        }
        mma_chunk(av, c, khiB, s_ow, acc);
    }

    // D layout: lane 0 holds column N=0 for rows 0..7 (VGPR v -> M=v),
    // lane 16 holds column N=0 for rows 8..15 (VGPR v -> M=v+8).
    if (lane == 0) {
        #pragma unroll
        for (int rI = 0; rI < 8; ++rI) out[rbase + rI] = acc[rI] + bias;
    } else if (lane == 16) {
        #pragma unroll
        for (int rI = 0; rI < 8; ++rI) out[rbase + 8 + rI] = acc[rI] + bias;
    }
}

extern "C" void kernel_launch(void* const* d_in, const int* in_sizes, int n_in,
                              void* d_out, int out_size, void* d_ws, size_t ws_size,
                              hipStream_t stream) {
    const float* t     = (const float*)d_in[0];
    const float* sin_w = (const float*)d_in[1];
    const float* sin_b = (const float*)d_in[2];
    const float* j2w   = (const float*)d_in[3];
    const float* j2b   = (const float*)d_in[4];
    const float* j4w   = (const float*)d_in[5];
    const float* j4b   = (const float*)d_in[6];
    const float* y2w   = (const float*)d_in[7];
    const float* y2b   = (const float*)d_in[8];
    const float* y4w   = (const float*)d_in[9];
    const float* y4b   = (const float*)d_in[10];
    const float* ow    = (const float*)d_in[11];
    const float* ob    = (const float*)d_in[12];
    float* out         = (float*)d_out;

    const int nrows  = in_sizes[0] / 2;   // B = 262144
    const int blocks = nrows / 128;       // 16 rows/wave * 8 waves/block

    FCNN_heteroBessel_kernel<<<blocks, 256, 0, stream>>>(
        t, sin_w, sin_b, j2w, j2b, j4w, j4b, y2w, y2b, y4w, y4b, ow, ob, out);
}
